// autoencoder_77919296684687
// MI455X (gfx1250) — matrix operand order repaired and checked by reading, not compiled, not run on hardware
//
#include <hip/hip_runtime.h>
#include <hip/hip_bf16.h>
#include <math.h>

typedef __attribute__((ext_vector_type(16))) _Float16 v16h;
typedef __attribute__((ext_vector_type(8)))  _Float16 v8h;
typedef __attribute__((ext_vector_type(8)))  float    v8f;
typedef __attribute__((ext_vector_type(4)))  float    v4f;

// ==================================================================
// Pre-swizzled f16 B-matrix layout (exact WMMA 16x16x32 B fragment):
//   fragment f = kb * nCT + ct   (kb = k-block of 32, ct = 16-col tile)
//   within fragment: lane L owns halves [f*512 + L*16 .. +15]
//     col = ct*16 + (L&15),  k = kb*32 + (L>>4)*8 + (i&7) + (i>>3)*16
// One wave loads its whole fragment with a single 32B v16h load.
// ==================================================================

// ------------------------------------------------------------------
// small prep kernels
// ------------------------------------------------------------------
__global__ void zero_f32(float* __restrict__ p, long n) {
  long i = (long)blockIdx.x * blockDim.x + threadIdx.x;
  if (i < n) p[i] = 0.f;
}

// Swizzled combined B for one RGCN layer: logical [K x 288],
// cols 0..255 = basis-combined W_r, cols 256..287 = loop_w.  nCT = 18.
__global__ void prep_bmat_sw(const float* __restrict__ V, const float* __restrict__ comb,
                             const float* __restrict__ loopw, _Float16* __restrict__ Bsw, int K) {
  int t = blockIdx.x * blockDim.x + threadIdx.x;
  int total = (K >> 5) * 18 * 512;
  if (t >= total) return;
  int f = t >> 9, q = t & 511;
  int L = q >> 4, i = q & 15;
  int kb = f / 18, ct = f % 18;
  int col = ct * 16 + (L & 15);
  int k = kb * 32 + ((L >> 4) << 3) + (i & 7) + ((i >> 3) << 4);
  float val;
  if (col < 256) {
    int r = col >> 5, o = col & 31;
    val = comb[r * 2 + 0] * V[(long)k * 32 + o] + comb[r * 2 + 1] * V[(long)K * 32 + (long)k * 32 + o];
  } else {
    val = loopw[(long)k * 32 + (col - 256)];
  }
  Bsw[t] = (_Float16)val;
}

// Swizzled MLP weights: W1sw logical [96 x 64] (nCT=4, 12 frags),
// W2sw two mats (mean,std) each [32 x 128] (nCT=8, 8 frags each), b1cat f32[64].
__global__ void prep_mlp_sw(const float* __restrict__ Wm1, const float* __restrict__ bm1,
                            const float* __restrict__ Ws1, const float* __restrict__ bs1,
                            const float* __restrict__ Wm2, const float* __restrict__ Ws2,
                            _Float16* __restrict__ W1sw, float* __restrict__ b1cat,
                            _Float16* __restrict__ W2sw) {
  int t = blockIdx.x * blockDim.x + threadIdx.x;
  if (t < 12 * 512) {                       // W1sw
    int f = t >> 9, q = t & 511;
    int L = q >> 4, i = q & 15;
    int kb = f >> 2, ct = f & 3;
    int col = ct * 16 + (L & 15);
    int k = kb * 32 + ((L >> 4) << 3) + (i & 7) + ((i >> 3) << 4);
    W1sw[t] = (_Float16)(col < 32 ? Wm1[k * 32 + col] : Ws1[k * 32 + (col - 32)]);
  }
  if (t < 64) b1cat[t] = (t < 32) ? bm1[t] : bs1[t - 32];
  if (t < 16 * 512) {                       // W2sw
    int f = t >> 9, q = t & 511;
    int L = q >> 4, i = q & 15;
    int mat = f >> 3, ct = f & 7;
    int col = ct * 16 + (L & 15);
    int k = ((L >> 4) << 3) + (i & 7) + ((i >> 3) << 4);
    W2sw[t] = (_Float16)(mat == 0 ? Wm2[k * 128 + col] : Ws2[k * 128 + col]);
  }
}

// ------------------------------------------------------------------
// A-fragment loaders (16-bit A 16x32 layout)
// ------------------------------------------------------------------
__device__ __forceinline__ v16h load_a_f32(const float* ap) {
  // ap points at A[row*K + k0 + hi*8]; halves come from ap[0..7], ap[16..23]
  v4f l0 = *(const v4f*)(ap);
  v4f l1 = *(const v4f*)(ap + 4);
  v4f l2 = *(const v4f*)(ap + 16);
  v4f l3 = *(const v4f*)(ap + 20);
  v16h a;
#pragma unroll
  for (int i = 0; i < 4; ++i) {
    a[i]      = (_Float16)l0[i];
    a[4 + i]  = (_Float16)l1[i];
    a[8 + i]  = (_Float16)l2[i];
    a[12 + i] = (_Float16)l3[i];
  }
  return a;
}

__device__ __forceinline__ v16h load_a_f16(const _Float16* ap) {
  v8h lo = *(const v8h*)(ap);
  v8h hi = *(const v8h*)(ap + 16);
  v16h a;
#pragma unroll
  for (int i = 0; i < 8; ++i) { a[i] = lo[i]; a[8 + i] = hi[i]; }
  return a;
}

// ------------------------------------------------------------------
// WMMA GEMM: C[N x nCT*16] (f16 out) = A[N x K] x Bsw, one wave/tile
// ------------------------------------------------------------------
template <bool A_IS_F16>
__global__ __launch_bounds__(256) void gemm_hw(const void* __restrict__ Av, int K,
                                               const _Float16* __restrict__ Bsw,
                                               _Float16* __restrict__ C, int nRows, int nCT) {
  int gtid = blockIdx.x * blockDim.x + threadIdx.x;
  int wave = gtid >> 5, lane = gtid & 31;
  int tile_m = wave / nCT, tile_n = wave % nCT;
  int m0 = tile_m * 16;
  if (m0 >= nRows) return;
  int colg = tile_n * 16 + (lane & 15);
  int row  = m0 + (lane & 15);
  int hi   = lane >> 4;
  int ldc  = nCT * 16;
  v8f acc = {};
  for (int k0 = 0; k0 < K; k0 += 32) {
    v16h a;
    if (A_IS_F16) a = load_a_f16((const _Float16*)Av + (long)row * K + k0 + (hi << 3));
    else          a = load_a_f32((const float*)Av   + (long)row * K + k0 + (hi << 3));
    v16h b = *(const v16h*)(Bsw + ((long)(k0 >> 5) * nCT + tile_n) * 512 + lane * 16);
    acc = __builtin_amdgcn_wmma_f32_16x16x32_f16(false, a, false, b, (short)0, acc, false, false);
  }
  int rowbase = m0 + (hi << 3);
#pragma unroll
  for (int j = 0; j < 8; ++j)
    C[(long)(rowbase + j) * ldc + colg] = (_Float16)acc[j];
}

// ------------------------------------------------------------------
// Edge gather + scatter-add: 4 edges per wave, unrolled so 4 gather
// lines are in flight before the atomics issue.  hw is f16 (L2-resident),
// accumulation in f32 via hardware global_atomic_add_f32.
// ------------------------------------------------------------------
__global__ __launch_bounds__(256) void edge_scatter(const int* __restrict__ src,
                                                    const int* __restrict__ dst,
                                                    const int* __restrict__ etype,
                                                    const _Float16* __restrict__ hwh,
                                                    float* __restrict__ agg, int E) {
  long t = (long)blockIdx.x * blockDim.x + threadIdx.x;
  int wave = (int)(t >> 5), lane = (int)(t & 31);
  int ebase = wave * 4;
  float v[4];
  int d[4];
  int cnt = 0;
#pragma unroll
  for (int i = 0; i < 4; ++i) {
    int e = ebase + i;
    if (e < E) {
      int s = src[e], r = etype[e];
      d[i] = dst[e];
      v[i] = (float)hwh[(long)s * 288 + r * 32 + lane];
      cnt = i + 1;
    }
  }
#pragma unroll
  for (int i = 0; i < 4; ++i)
    if (i < cnt) unsafeAtomicAdd(&agg[(long)d[i] * 32 + lane], v[i]);
}

__global__ void finalize_layer(const float* __restrict__ agg, const _Float16* __restrict__ hwh,
                               const float* __restrict__ bias, _Float16* __restrict__ hout, int N) {
  long t = (long)blockIdx.x * blockDim.x + threadIdx.x;
  if (t >= (long)N * 32) return;
  long n = t >> 5;
  int c = (int)(t & 31);
  float v = agg[t] + (float)hwh[n * 288 + 256 + c] + bias[c];
  hout[t] = (_Float16)tanhf(v);
}

// ------------------------------------------------------------------
// MLP stage 1: Hcat[N x 64] f16 = relu([h0|h1|h2][N x 96] @ W1 + b1)
// ------------------------------------------------------------------
__global__ __launch_bounds__(256) void mlp_stage1(const _Float16* __restrict__ h0,
                                                  const _Float16* __restrict__ h1,
                                                  const _Float16* __restrict__ h2,
                                                  const _Float16* __restrict__ W1sw,
                                                  const float* __restrict__ b1,
                                                  _Float16* __restrict__ Hcat, int nRows) {
  int gtid = blockIdx.x * blockDim.x + threadIdx.x;
  int wave = gtid >> 5, lane = gtid & 31;
  int tile_m = wave >> 2, tile_n = wave & 3;
  int m0 = tile_m * 16;
  if (m0 >= nRows) return;
  int colg = tile_n * 16 + (lane & 15);
  int row  = m0 + (lane & 15);
  int hi   = lane >> 4;
  const _Float16* hs[3] = {h0, h1, h2};
  v8f acc = {};
#pragma unroll
  for (int kb = 0; kb < 3; ++kb) {
    v16h a = load_a_f16(hs[kb] + (long)row * 32 + (hi << 3));
    v16h b = *(const v16h*)(W1sw + ((long)(kb * 4 + tile_n)) * 512 + lane * 16);
    acc = __builtin_amdgcn_wmma_f32_16x16x32_f16(false, a, false, b, (short)0, acc, false, false);
  }
  float bb = b1[colg];
  int rowbase = m0 + (hi << 3);
#pragma unroll
  for (int j = 0; j < 8; ++j) {
    float v = acc[j] + bb;
    Hcat[(long)(rowbase + j) * 64 + colg] = (_Float16)(v > 0.f ? v : 0.f);
  }
}

// ------------------------------------------------------------------
// MLP stage 2 + reparameterization: z = mean + noise * exp(log_std)
// ------------------------------------------------------------------
__global__ __launch_bounds__(256) void mlp_stage2(const _Float16* __restrict__ Hcat,
                                                  const _Float16* __restrict__ W2sw,
                                                  const float* __restrict__ bm2,
                                                  const float* __restrict__ bs2,
                                                  const float* __restrict__ noise,
                                                  float* __restrict__ z, int nRows) {
  int gtid = blockIdx.x * blockDim.x + threadIdx.x;
  int wave = gtid >> 5, lane = gtid & 31;
  int tile_m = wave >> 3, tile_n = wave & 7;
  int m0 = tile_m * 16;
  if (m0 >= nRows) return;
  int colg = tile_n * 16 + (lane & 15);
  int row  = m0 + (lane & 15);
  int hi   = lane >> 4;

  const _Float16* ap = Hcat + (long)row * 64 + (hi << 3);
  v16h a_m = load_a_f16(ap);
  v16h a_s = load_a_f16(ap + 32);
  v16h b_m = *(const v16h*)(W2sw + ((long)tile_n) * 512 + lane * 16);
  v16h b_s = *(const v16h*)(W2sw + ((long)(8 + tile_n)) * 512 + lane * 16);

  v8f c_m = {}, c_s = {};
  c_m = __builtin_amdgcn_wmma_f32_16x16x32_f16(false, a_m, false, b_m, (short)0, c_m, false, false);
  c_s = __builtin_amdgcn_wmma_f32_16x16x32_f16(false, a_s, false, b_s, (short)0, c_s, false, false);

  float bmv = bm2[colg], bsv = bs2[colg];
  int rowbase = m0 + (hi << 3);
#pragma unroll
  for (int j = 0; j < 8; ++j) {
    long idx = (long)(rowbase + j) * 128 + colg;
    z[idx] = (c_m[j] + bmv) + noise[idx] * __expf(c_s[j] + bsv);
  }
}

// ------------------------------------------------------------------
// Output readouts (b128 vector loads)
// ------------------------------------------------------------------
__global__ void gather_results(const float* __restrict__ z, const int* __restrict__ user_idx,
                               const int* __restrict__ item_idx, float* __restrict__ out, int U) {
  long t = (long)blockIdx.x * blockDim.x + threadIdx.x;   // one float4 per thread
  if (t >= (long)2 * U * 32) return;
  int rowp = (int)(t >> 5), c4 = (int)(t & 31);
  int idx = rowp < U ? user_idx[rowp] : item_idx[rowp - U];
  ((v4f*)out)[t] = ((const v4f*)z)[(long)idx * 32 + c4];
}

__global__ __launch_bounds__(256) void pair_dots(const float* __restrict__ z,
                                                 const int* __restrict__ user_idx,
                                                 const int* __restrict__ item_idx,
                                                 const int* __restrict__ te_src,
                                                 const int* __restrict__ te_dst,
                                                 float* __restrict__ out, int U, int T) {
  long t = (long)blockIdx.x * blockDim.x + threadIdx.x;
  int p = (int)(t >> 5), lane = (int)(t & 31);
  if (p >= U + T) return;
  int a, b;
  if (p < U) { a = user_idx[p]; b = item_idx[p]; }
  else       { a = te_src[p - U]; b = te_dst[p - U]; }
  v4f pa = ((const v4f*)z)[(long)a * 32 + lane];
  v4f pb = ((const v4f*)z)[(long)b * 32 + lane];
  v4f pr = pa * pb;
  float s = pr[0] + pr[1] + pr[2] + pr[3];
#pragma unroll
  for (int m = 16; m >= 1; m >>= 1) s += __shfl_xor(s, m, 32);
  if (lane == 0) out[p] = s;
}

// ------------------------------------------------------------------
extern "C" void kernel_launch(void* const* d_in, const int* in_sizes, int n_in,
                              void* d_out, int out_size, void* d_ws, size_t ws_size,
                              hipStream_t stream) {
  const float* x        = (const float*)d_in[0];
  const int*   src      = (const int*)d_in[1];
  const int*   dst      = (const int*)d_in[2];
  const int*   etype    = (const int*)d_in[3];
  const float* noise    = (const float*)d_in[4];
  const int*   user_idx = (const int*)d_in[5];
  const int*   item_idx = (const int*)d_in[6];
  const int*   te_src   = (const int*)d_in[7];
  const int*   te_dst   = (const int*)d_in[8];
  const float* Vs[3]    = {(const float*)d_in[9],  (const float*)d_in[13], (const float*)d_in[17]};
  const float* combs[3] = {(const float*)d_in[10], (const float*)d_in[14], (const float*)d_in[18]};
  const float* loops[3] = {(const float*)d_in[11], (const float*)d_in[15], (const float*)d_in[19]};
  const float* bls[3]   = {(const float*)d_in[12], (const float*)d_in[16], (const float*)d_in[20]};
  const float* Wm1 = (const float*)d_in[21]; const float* bm1 = (const float*)d_in[22];
  const float* Wm2 = (const float*)d_in[23]; const float* bm2 = (const float*)d_in[24];
  const float* Ws1 = (const float*)d_in[25]; const float* bs1 = (const float*)d_in[26];
  const float* Ws2 = (const float*)d_in[27]; const float* bs2 = (const float*)d_in[28];

  const int N = in_sizes[0] / 128;
  const int E = in_sizes[1];
  const int U = in_sizes[5];
  const int T = in_sizes[7];

  // Workspace layout (f16 hw keeps the gather working set L2-resident)
  char* ws = (char*)d_ws;
  _Float16* hwh = (_Float16*)ws;                                   // N*288 f16 (115 MB)
  float*    agg = (float*)(ws + (size_t)N * 288 * 2);              // N*32  f32
  _Float16* h0  = (_Float16*)(ws + (size_t)N * 288 * 2 + (size_t)N * 32 * 4);
  _Float16* h1  = h0 + (size_t)N * 32;                             // f16
  _Float16* h2  = h1 + (size_t)N * 32;
  float*    z   = (float*)(h2 + (size_t)N * 32);                   // N*128 f32
  _Float16* Bsw  = (_Float16*)(z + (size_t)N * 128);               // 4*18*512 f16
  _Float16* W1sw = Bsw + 4 * 18 * 512;
  _Float16* W2sw = W1sw + 12 * 512;
  float*    b1cat = (float*)(W2sw + 16 * 512);
  _Float16* Hcat = hwh;     // alias: hw dead after layer-2 finalize

  const int tpb = 256;
  const void* hin = x;
  int K = 128;
  _Float16* houts[3] = {h0, h1, h2};
  long rowTiles = (N + 15) / 16;

  for (int l = 0; l < 3; ++l) {
    int bmTotal = (K >> 5) * 18 * 512;
    prep_bmat_sw<<<(unsigned)((bmTotal + tpb - 1) / tpb), tpb, 0, stream>>>(
        Vs[l], combs[l], loops[l], Bsw, K);
    zero_f32<<<(unsigned)(((long)N * 32 + tpb - 1) / tpb), tpb, 0, stream>>>(agg, (long)N * 32);
    long tiles = rowTiles * 18;
    if (l == 0)
      gemm_hw<false><<<(unsigned)((tiles * 32 + tpb - 1) / tpb), tpb, 0, stream>>>(
          hin, K, Bsw, hwh, N, 18);
    else
      gemm_hw<true><<<(unsigned)((tiles * 32 + tpb - 1) / tpb), tpb, 0, stream>>>(
          hin, K, Bsw, hwh, N, 18);
    long escWaves = ((long)E + 3) / 4;
    edge_scatter<<<(unsigned)((escWaves * 32 + tpb - 1) / tpb), tpb, 0, stream>>>(
        src, dst, etype, hwh, agg, E);
    finalize_layer<<<(unsigned)(((long)N * 32 + tpb - 1) / tpb), tpb, 0, stream>>>(
        agg, hwh, bls[l], houts[l], N);
    hin = houts[l];
    K = 32;
  }

  prep_mlp_sw<<<(unsigned)((16 * 512 + tpb - 1) / tpb), tpb, 0, stream>>>(
      Wm1, bm1, Ws1, bs1, Wm2, Ws2, W1sw, b1cat, W2sw);
  mlp_stage1<<<(unsigned)((rowTiles * 4 * 32 + tpb - 1) / tpb), tpb, 0, stream>>>(
      h0, h1, h2, W1sw, b1cat, Hcat, N);
  mlp_stage2<<<(unsigned)((rowTiles * 8 * 32 + tpb - 1) / tpb), tpb, 0, stream>>>(
      Hcat, W2sw, bm2, bs2, noise, z, N);

  float* out = (float*)d_out;
  gather_results<<<(unsigned)(((long)2 * U * 32 + tpb - 1) / tpb), tpb, 0, stream>>>(
      z, user_idx, item_idx, out + U + T, U);
  pair_dots<<<(unsigned)(((long)(U + T) * 32 + tpb - 1) / tpb), tpb, 0, stream>>>(
      z, user_idx, item_idx, te_src, te_dst, out, U, T);
}
